// CausalSelfAttention_6828998001256
// MI455X (gfx1250) — matrix operand order repaired and checked by reading, not compiled, not run on hardware
//
#include <hip/hip_runtime.h>
#include <cstddef>

// ---------------------------------------------------------------------------
// CDNA5 / gfx1250 fused causal self-attention, bf16 WMMA (f32 accumulate),
// TDM (tensor_load_to_lds) for K-tile staging in the flash-attention kernel.
// B=2, T=2048, C=1024, H=16, DH=64.
// ---------------------------------------------------------------------------

#define Bn  2
#define Tn  2048
#define Cn  1024
#define Hn  16
#define DHn 64
#define NEGBIG (-3.0e38f)

typedef __attribute__((ext_vector_type(16))) __bf16        v16bf;
typedef __attribute__((ext_vector_type(8)))  float         v8f;
typedef __attribute__((ext_vector_type(4)))  unsigned int  v4u;
typedef __attribute__((ext_vector_type(4)))  int           v4i;
typedef __attribute__((ext_vector_type(8)))  int           v8i;

union Frag {
    v16bf v;
    uint4 q[2];
    __bf16 e[16];
};

static __device__ __forceinline__ v8f wmma_bf16(v16bf a, v16bf b, v8f c) {
    // (neg_a, A, neg_b, B, c_mod, C, reuse_a, reuse_b)
    return __builtin_amdgcn_wmma_f32_16x16x32_bf16(false, a, false, b, (short)0, c,
                                                   false, false);
}

// Issue a TDM 2D tile load: bf16 tile [tile_d1 rows][tile_d0 cols] from a
// row-major global region with stride `stride0` (elements) into LDS at
// byte address `lds_addr` (packed row-major).  D# layout per
// CDNA5 ISA.  Tracked with TENSORcnt.
static __device__ __forceinline__ void tdm_load_tile_bf16(
    unsigned lds_addr, const void* gptr, int tile_d0, int tile_d1, int stride0) {
    unsigned long long ga = (unsigned long long)(size_t)gptr;
    v4u g0;
    g0[0] = 1u;                                        // count=1, user mode
    g0[1] = lds_addr;                                  // lds_addr [63:32]
    g0[2] = (unsigned)(ga & 0xFFFFFFFFu);              // global_addr lo
    g0[3] = (unsigned)((ga >> 32) & 0x01FFFFFFu) | (2u << 30);  // type=2
    v8i g1;
    g1[0] = 0x00010000;                                // data_size=1 (2 bytes)
    g1[1] = (tile_d0 & 0xFFFF) << 16;                  // tensor_dim0[15:0]
    g1[2] = (tile_d1 & 0xFFFF) << 16;                  // tensor_dim1[15:0]
    g1[3] = (tile_d0 & 0xFFFF) << 16;                  // tile_dim0
    g1[4] = (tile_d1 & 0xFFFF);                        // tile_dim1 (tile_dim2=0)
    g1[5] = stride0;                                   // tensor_dim0_stride lo
    g1[6] = 0;
    g1[7] = 0;
    v4i g2 = {};                                       // 2D: groups 2/3 unused
    v4i g3 = {};
#if defined(__clang_major__) && (__clang_major__ >= 23)
    v8i g4 = {};
    __builtin_amdgcn_tensor_load_to_lds(g0, g1, g2, g3, g4, 0);
#else
    __builtin_amdgcn_tensor_load_to_lds(g0, g1, g2, g3, 0);
#endif
}

// -------------------------------- converters -------------------------------

__global__ void cvt_f32_to_bf16(const float* __restrict__ in,
                                __bf16* __restrict__ out, int n) {
    int i = blockIdx.x * blockDim.x + threadIdx.x;
    if (i < n) out[i] = (__bf16)in[i];
}

// out[n*Cn + k] = (bf16) in[k*Cn + n]   (transpose 1024x1024 weight)
__global__ void cvt_transpose_w(const float* __restrict__ in,
                                __bf16* __restrict__ out) {
    int i = blockIdx.x * blockDim.x + threadIdx.x;   // 0 .. Cn*Cn-1
    int n = i >> 10;
    int k = i & (Cn - 1);
    out[i] = (__bf16)in[(size_t)k * Cn + n];
}

// ------------------------------ projection GEMM ----------------------------
// C[M=B*T][N=1024] = A_bf16[M][1024] @ W + bias, with W given transposed
// (BT[n][k]).  One wave computes a 16x64 tile (4 WMMA accumulators, A-frag
// reused 4x); 4 waves per block -> block tile 16x256.
// mode 0: V   -> bf16 scatter to [B,H,T,DH]
// mode 1: Q/K -> RoPE then bf16 scatter to [B,H,T,DH]
// mode 2: out -> f32 row-major [B*T][C]
__global__ __launch_bounds__(128) void gemm_bf16(
    const __bf16* __restrict__ A, const __bf16* __restrict__ BT,
    const float* __restrict__ bias, void* __restrict__ out,
    const float* __restrict__ fcos, const float* __restrict__ fsin, int mode) {

    const int wave = threadIdx.x >> 5;
    const int lane = threadIdx.x & 31;
    const int half = lane >> 4;
    const int ln   = lane & 15;

    const int row0    = blockIdx.x * 16;                    // 16-row tile
    const int colbase = (blockIdx.y * 4 + wave) * 64;       // 64-col tile

    const __bf16* arow = A + (size_t)(row0 + ln) * Cn;
    const __bf16* brow0 = BT + (size_t)(colbase +  0 + ln) * Cn;
    const __bf16* brow1 = BT + (size_t)(colbase + 16 + ln) * Cn;
    const __bf16* brow2 = BT + (size_t)(colbase + 32 + ln) * Cn;
    const __bf16* brow3 = BT + (size_t)(colbase + 48 + ln) * Cn;

    v8f acc0 = {}, acc1 = {}, acc2 = {}, acc3 = {};
    #pragma unroll 4
    for (int k0 = 0; k0 < Cn; k0 += 32) {
        Frag a, b0, b1, b2, b3;
        // A 16x32 bf16 fragment: lane row = ln; elem i -> K=16*half+i
        a.q[0] = *(const uint4*)(arow + k0 + 16 * half);
        a.q[1] = *(const uint4*)(arow + k0 + 16 * half + 8);
        // B 32x16 bf16 fragments: lane col = ln; elem i -> K=16*half+i
        b0.q[0] = *(const uint4*)(brow0 + k0 + 16 * half);
        b0.q[1] = *(const uint4*)(brow0 + k0 + 16 * half + 8);
        b1.q[0] = *(const uint4*)(brow1 + k0 + 16 * half);
        b1.q[1] = *(const uint4*)(brow1 + k0 + 16 * half + 8);
        b2.q[0] = *(const uint4*)(brow2 + k0 + 16 * half);
        b2.q[1] = *(const uint4*)(brow2 + k0 + 16 * half + 8);
        b3.q[0] = *(const uint4*)(brow3 + k0 + 16 * half);
        b3.q[1] = *(const uint4*)(brow3 + k0 + 16 * half + 8);
        acc0 = wmma_bf16(a.v, b0.v, acc0);
        acc1 = wmma_bf16(a.v, b1.v, acc1);
        acc2 = wmma_bf16(a.v, b2.v, acc2);
        acc3 = wmma_bf16(a.v, b3.v, acc3);
    }

    v8f accs[4] = {acc0, acc1, acc2, acc3};
    #pragma unroll
    for (int ct = 0; ct < 4; ++ct) {
        const int col0 = colbase + ct * 16;
        const float bn = bias[col0 + ln];
        v8f acc = accs[ct];
        #pragma unroll
        for (int j = 0; j < 8; ++j) acc[j] += bn;

        if (mode == 2) {
            float* o = (float*)out;
            #pragma unroll
            for (int j = 0; j < 8; ++j) {
                int rg = row0 + j + 8 * half;           // D row = j + 8*half
                o[(size_t)rg * Cn + col0 + ln] = acc[j];
            }
            continue;
        }

        const int cg = col0 + ln;        // channel 0..1023
        const int h  = cg >> 6;
        const int d  = cg & 63;
        __bf16* ob = (__bf16*)out;

        if (mode == 1) {
            const int  f    = d >> 1;    // 0..31
            const bool even = (d & 1) == 0;
            #pragma unroll
            for (int j = 0; j < 8; ++j) {
                int rg = row0 + j + 8 * half;
                int t  = rg & (Tn - 1);
                int b_ = rg >> 11;
                float c = fcos[t * 32 + f];
                float s = fsin[t * 32 + f];
                float other = __shfl_xor(acc[j], 1, 32);   // pair channel
                float val = even ? (acc[j] * c - other * s)
                                 : (other * s + acc[j] * c);
                ob[(((size_t)(b_ * Hn + h)) * Tn + t) * DHn + d] = (__bf16)val;
            }
        } else {
            #pragma unroll
            for (int j = 0; j < 8; ++j) {
                int rg = row0 + j + 8 * half;
                int t  = rg & (Tn - 1);
                int b_ = rg >> 11;
                ob[(((size_t)(b_ * Hn + h)) * Tn + t) * DHn + d] = (__bf16)acc[j];
            }
        }
    }
}

// ------------------------------ flash attention ----------------------------
// grid: (T/64, B*H), block 128 (4 waves).  Each wave owns 16 query rows.
// Streams keys in blocks of 32 with online softmax.  K tile is DMA'd into
// LDS by the Tensor Data Mover (TENSORcnt); V tile is staged transposed by
// the block; P round-trips through per-wave LDS to convert the C/D fragment
// layout into the A fragment layout.
__global__ __launch_bounds__(128) void flash_attn(
    const __bf16* __restrict__ qr, const __bf16* __restrict__ kr,
    const __bf16* __restrict__ vb, __bf16* __restrict__ attn) {

    __shared__ __align__(16) __bf16 Kt[32 * 64];        // [key][d]  (TDM dest)
    __shared__ __align__(16) __bf16 VT[64 * 32];        // [d][key]
    __shared__ __align__(16) __bf16 Pl[4][16 * 32];     // per-wave P tile

    const int tid  = threadIdx.x;
    const int wave = tid >> 5;
    const int lane = tid & 31;
    const int half = lane >> 4;
    const int ln   = lane & 15;

    const int qt = blockIdx.x;          // 64-row query tile
    const int bh = blockIdx.y;          // b*H + h
    const int b_ = bh >> 4;
    const int h  = bh & 15;

    const unsigned ktLds = (unsigned)(size_t)(&Kt[0]);  // LDS byte address

    // Q A-fragments (d 0..31 and 32..63), rows = qt*64 + wave*16 + ln
    const __bf16* qbase = qr + ((size_t)bh * Tn + qt * 64 + wave * 16) * DHn;
    Frag aq0, aq1;
    {
        const __bf16* rowp = qbase + (size_t)ln * DHn;
        aq0.q[0] = *(const uint4*)(rowp + 16 * half);
        aq0.q[1] = *(const uint4*)(rowp + 16 * half + 8);
        aq1.q[0] = *(const uint4*)(rowp + 32 + 16 * half);
        aq1.q[1] = *(const uint4*)(rowp + 32 + 16 * half + 8);
    }

    float m[8], l[8];
    v8f o0 = {}, o1 = {}, o2 = {}, o3 = {};
    #pragma unroll
    for (int j = 0; j < 8; ++j) { m[j] = NEGBIG; l[j] = 0.f; }

    const __bf16* kbase = kr + (size_t)bh * Tn * DHn;
    const __bf16* vbase = vb + (size_t)bh * Tn * DHn;
    const int kbEnd = 2 * qt + 1;       // inclusive; 32 keys per block

    for (int kb = 0; kb <= kbEnd; ++kb) {
        // K block 32x64 via Tensor Data Mover (one wave issues the DMA)
        if (wave == 0) {
            tdm_load_tile_bf16(ktLds, kbase + (size_t)kb * 32 * DHn,
                               /*d0=*/64, /*d1=*/32, /*stride0=*/64);
        }
        if (kb < kbEnd) {               // prefetch next V tile (uniform)
            __builtin_prefetch(vbase + (size_t)(kb + 1) * 32 * DHn, 0, 0);
        }
        // cooperative load: V block transposed into VT[d][key]
        {
            const __bf16* src = vbase + (size_t)kb * 32 * DHn;
            int kk   = tid >> 2;             // 0..31
            int dseg = (tid & 3) * 16;       // 0,16,32,48
            Frag t;
            t.q[0] = *(const uint4*)(src + kk * 64 + dseg);
            t.q[1] = *(const uint4*)(src + kk * 64 + dseg + 8);
            #pragma unroll
            for (int j = 0; j < 16; ++j) VT[(dseg + j) * 32 + kk] = t.e[j];
        }
        if (wave == 0) __builtin_amdgcn_s_wait_tensorcnt(0);  // K tile landed
        __syncthreads();

        // S = Q @ K^T  (two key sub-tiles of 16, contraction over d=64)
        v8f s0 = {}, s1 = {};
        #pragma unroll
        for (int db = 0; db < 64; db += 32) {
            Frag bk0, bk1;
            const __bf16* p0 = &Kt[ln * 64 + db + 16 * half];
            bk0.q[0] = *(const uint4*)(p0);
            bk0.q[1] = *(const uint4*)(p0 + 8);
            const __bf16* p1 = &Kt[(16 + ln) * 64 + db + 16 * half];
            bk1.q[0] = *(const uint4*)(p1);
            bk1.q[1] = *(const uint4*)(p1 + 8);
            v16bf aq = db ? aq1.v : aq0.v;
            s0 = wmma_bf16(aq, bk0.v, s0);
            s1 = wmma_bf16(aq, bk1.v, s1);
        }

        // scale + causal mask + online softmax statistics
        const int tq0 = qt * 64 + wave * 16 + 8 * half;   // + j = row
        const int k0g = kb * 32 + ln;
        float mnew[8], alpha[8];
        #pragma unroll
        for (int j = 0; j < 8; ++j) {
            float a0 = s0[j] * 0.125f;
            float a1 = s1[j] * 0.125f;
            int row = tq0 + j;
            if (k0g      > row) a0 = NEGBIG;
            if (k0g + 16 > row) a1 = NEGBIG;
            s0[j] = a0; s1[j] = a1;
            float mx = fmaxf(a0, a1);
            mx = fmaxf(mx, __shfl_xor(mx, 1, 32));
            mx = fmaxf(mx, __shfl_xor(mx, 2, 32));
            mx = fmaxf(mx, __shfl_xor(mx, 4, 32));
            mx = fmaxf(mx, __shfl_xor(mx, 8, 32));
            mnew[j] = fmaxf(m[j], mx);
        }
        #pragma unroll
        for (int j = 0; j < 8; ++j) {
            float p0 = __expf(s0[j] - mnew[j]);
            float p1 = __expf(s1[j] - mnew[j]);
            s0[j] = p0; s1[j] = p1;
            float rs = p0 + p1;
            rs += __shfl_xor(rs, 1, 32);
            rs += __shfl_xor(rs, 2, 32);
            rs += __shfl_xor(rs, 4, 32);
            rs += __shfl_xor(rs, 8, 32);
            alpha[j] = __expf(m[j] - mnew[j]);
            l[j] = l[j] * alpha[j] + rs;
            m[j] = mnew[j];
        }
        #pragma unroll
        for (int j = 0; j < 8; ++j) {
            o0[j] *= alpha[j]; o1[j] *= alpha[j];
            o2[j] *= alpha[j]; o3[j] *= alpha[j];
        }

        // P (C-layout) -> per-wave LDS -> A-layout fragment
        __bf16* pl = &Pl[wave][0];
        #pragma unroll
        for (int j = 0; j < 8; ++j) {
            pl[(j + 8 * half) * 32 + ln]      = (__bf16)s0[j];
            pl[(j + 8 * half) * 32 + 16 + ln] = (__bf16)s1[j];
        }
        Frag ap;   // LDS ds ops are in-order within a wave: no barrier needed
        {
            const __bf16* rowp = pl + ln * 32;
            ap.q[0] = *(const uint4*)(rowp + 16 * half);
            ap.q[1] = *(const uint4*)(rowp + 16 * half + 8);
        }

        // O += P @ V  (4 d-tiles of 16, contraction over 32 keys)
        {
            Frag bv;
            const __bf16* p;
            p = &VT[(0 * 16 + ln) * 32 + 16 * half];
            bv.q[0] = *(const uint4*)p; bv.q[1] = *(const uint4*)(p + 8);
            o0 = wmma_bf16(ap.v, bv.v, o0);
            p = &VT[(1 * 16 + ln) * 32 + 16 * half];
            bv.q[0] = *(const uint4*)p; bv.q[1] = *(const uint4*)(p + 8);
            o1 = wmma_bf16(ap.v, bv.v, o1);
            p = &VT[(2 * 16 + ln) * 32 + 16 * half];
            bv.q[0] = *(const uint4*)p; bv.q[1] = *(const uint4*)(p + 8);
            o2 = wmma_bf16(ap.v, bv.v, o2);
            p = &VT[(3 * 16 + ln) * 32 + 16 * half];
            bv.q[0] = *(const uint4*)p; bv.q[1] = *(const uint4*)(p + 8);
            o3 = wmma_bf16(ap.v, bv.v, o3);
        }
        __syncthreads();   // protect Kt/VT before next iteration overwrites
    }

    // normalize and write to attn[B,T,C] (bf16, row-major)
    #pragma unroll
    for (int j = 0; j < 8; ++j) {
        float inv = 1.0f / l[j];
        int trow = qt * 64 + wave * 16 + 8 * half + j;
        __bf16* outp = attn + ((size_t)(b_ * Tn + trow)) * Cn + h * DHn + ln;
        outp[0]  = (__bf16)(o0[j] * inv);
        outp[16] = (__bf16)(o1[j] * inv);
        outp[32] = (__bf16)(o2[j] * inv);
        outp[48] = (__bf16)(o3[j] * inv);
    }
}

// --------------------------------- launcher --------------------------------

extern "C" void kernel_launch(void* const* d_in, const int* in_sizes, int n_in,
                              void* d_out, int out_size, void* d_ws, size_t ws_size,
                              hipStream_t stream) {
    (void)in_sizes; (void)n_in; (void)out_size; (void)ws_size;

    const float* x    = (const float*)d_in[0];
    // d_in[1] = mask (causality handled analytically)
    const float* fcos = (const float*)d_in[2];
    const float* fsin = (const float*)d_in[3];
    const float* wq   = (const float*)d_in[4];
    const float* bq   = (const float*)d_in[5];
    const float* wk   = (const float*)d_in[6];
    const float* bk   = (const float*)d_in[7];
    const float* wv   = (const float*)d_in[8];
    const float* bv   = (const float*)d_in[9];
    const float* wo   = (const float*)d_in[10];
    const float* bo   = (const float*)d_in[11];
    float* out = (float*)d_out;

    const size_t NX = (size_t)Bn * Tn * Cn;      // 4,194,304
    const size_t NW = (size_t)Cn * Cn;           // 1,048,576

    char* ws = (char*)d_ws;
    __bf16* xb   = (__bf16*)ws;  ws += NX * 2;
    __bf16* wqT  = (__bf16*)ws;  ws += NW * 2;
    __bf16* wkT  = (__bf16*)ws;  ws += NW * 2;
    __bf16* wvT  = (__bf16*)ws;  ws += NW * 2;
    __bf16* woT  = (__bf16*)ws;  ws += NW * 2;
    __bf16* qr   = (__bf16*)ws;  ws += NX * 2;   // [B,H,T,DH] rope'd
    __bf16* kr   = (__bf16*)ws;  ws += NX * 2;   // [B,H,T,DH] rope'd
    __bf16* vbuf = (__bf16*)ws;  ws += NX * 2;   // [B,H,T,DH]
    __bf16* attn = (__bf16*)ws;  ws += NX * 2;   // [B,T,C]

    // 1) converts
    cvt_f32_to_bf16<<<(int)(NX / 256), 256, 0, stream>>>(x, xb, (int)NX);
    cvt_transpose_w<<<(int)(NW / 256), 256, 0, stream>>>(wq, wqT);
    cvt_transpose_w<<<(int)(NW / 256), 256, 0, stream>>>(wk, wkT);
    cvt_transpose_w<<<(int)(NW / 256), 256, 0, stream>>>(wv, wvT);
    cvt_transpose_w<<<(int)(NW / 256), 256, 0, stream>>>(wo, woT);

    // 2) QKV projections (+RoPE for Q,K), scattered to [B,H,T,DH] bf16
    dim3 ggrid((Bn * Tn) / 16, Cn / 256);
    gemm_bf16<<<ggrid, 128, 0, stream>>>(xb, wqT, bq, qr,   fcos, fsin, 1);
    gemm_bf16<<<ggrid, 128, 0, stream>>>(xb, wkT, bk, kr,   fcos, fsin, 1);
    gemm_bf16<<<ggrid, 128, 0, stream>>>(xb, wvT, bv, vbuf, fcos, fsin, 0);

    // 3) fused causal flash attention -> attn [B,T,C] bf16
    flash_attn<<<dim3(Tn / 64, Bn * Hn), 128, 0, stream>>>(qr, kr, vbuf, attn);

    // 4) output projection -> f32
    gemm_bf16<<<ggrid, 128, 0, stream>>>(attn, woT, bo, out, fcos, fsin, 2);
}
